// RotDecSALayer_67327907332648
// MI455X (gfx1250) — compile-verified, matrix operand order checked by reading, not run on hardware
//
#include <hip/hip_runtime.h>
#include <hip/hip_bf16.h>
#include <math.h>

// ---------------------------------------------------------------------------
// Types matching the probe-confirmed WMMA / TDM builtin signatures
// ---------------------------------------------------------------------------
typedef __bf16 bf16_t;
typedef __attribute__((ext_vector_type(16))) __bf16 v16bf;
typedef __attribute__((ext_vector_type(8)))  __bf16 bf16x8;
typedef __attribute__((ext_vector_type(8)))  float  v8f;
typedef __attribute__((ext_vector_type(4)))  unsigned int v4u;
typedef __attribute__((ext_vector_type(8)))  int v8i;
typedef __attribute__((ext_vector_type(4)))  int v4i;

#define T_SEQ  2048
#define BATCH  2
#define CMODEL 1024
#define FDIM   4096
#define NHEAD  16
#define DHEAD  64
#define MROWS  (T_SEQ * BATCH)   // 4096 token rows, row = t*B + b

// ---------------------------------------------------------------------------
// LDS byte address of a __shared__ object: low 32 bits of the flat address
// (ISA 10.2: LDS aperture).
// ---------------------------------------------------------------------------
__device__ __forceinline__ unsigned lds_addr_of(const void* p) {
    return (unsigned)(unsigned long long)p;
}

// ---------------------------------------------------------------------------
// Tensor Data Mover: 2D tile load Global -> LDS.
// D# per ISA 8.3/8.4: group0 = {count|flags, lds_addr, global_addr lo,
// global_addr hi | type=2}; group1 packs data_size=2B, tensor/tile dims,
// dim0 stride. Groups 2/3 zero (tile_dim2 = 0 -> 2D). cpol = 0.
// This toolchain exposes the 6-arg builtin (extra int32x8 group).
// Rows land contiguously in LDS: [tile_d1][tile_d0] row-major.
// ---------------------------------------------------------------------------
__device__ __forceinline__ void tdm_load_2d(const bf16_t* gsrc, unsigned lds_byte,
                                            int tile_d0, int tile_d1, int stride) {
    unsigned long long ga = (unsigned long long)gsrc;
    v4u g0;
    g0[0] = 1u;                                           // count=1 (valid), user mode
    g0[1] = lds_byte;                                     // lds_addr (bytes)
    g0[2] = (unsigned)(ga & 0xFFFFFFFFu);                 // global_addr[31:0]
    g0[3] = (unsigned)((ga >> 32) & 0x01FFFFFFu) | (2u << 30);  // addr[56:32] | type=2
    const int TD1 = 1 << 20;                              // generous tensor_dim1
    v8i g1;
    g1[0] = (1 << 16);                                    // wg_mask=0, data_size=1 (2 bytes)
    g1[1] = (stride & 0xFFFF) << 16;                      // tensor_dim0[15:0]
    g1[2] = ((stride >> 16) & 0xFFFF) | ((TD1 & 0xFFFF) << 16);   // dim0 hi | dim1 lo
    g1[3] = ((TD1 >> 16) & 0xFFFF) | (tile_d0 << 16);     // dim1 hi | tile_dim0
    g1[4] = tile_d1 & 0xFFFF;                             // tile_dim1, tile_dim2=0
    g1[5] = stride;                                       // tensor_dim0_stride[31:0]
    g1[6] = 0;                                            // stride hi, dim1_stride lo
    g1[7] = 0;
    v4i z4 = {0, 0, 0, 0};
    v8i z8 = {0, 0, 0, 0, 0, 0, 0, 0};
    __builtin_amdgcn_tensor_load_to_lds(g0, g1, z4, z4, z8, 0);
}

// ---------------------------------------------------------------------------
// Fragment loader: 16-bit A/B WMMA operand layout from a row of 32 contiguous
// bf16 (ISA 7.12.2): lanes 0-15 take K 0-7/16-23, lanes 16-31 take K 8-15/24-31.
// ---------------------------------------------------------------------------
__device__ __forceinline__ v16bf load_frag16(const bf16_t* p, int lo) {
    bf16x8 a = *(const bf16x8*)(p + lo);
    bf16x8 b = *(const bf16x8*)(p + 16 + lo);
    v16bf r;
#pragma unroll
    for (int i = 0; i < 8; ++i) { r[i] = a[i]; r[i + 8] = b[i]; }
    return r;
}

// ---------------------------------------------------------------------------
// 16-lane butterfly reductions via v_permlane16_b32 (pure VALU, no DS pipe).
// Selector nibbles implement lane ^= {1,2,4,8} inside each 16-lane row.
// ---------------------------------------------------------------------------
__device__ __forceinline__ float permx16(float v, unsigned s0, unsigned s1) {
    unsigned u = __float_as_uint(v);
    return __uint_as_float(__builtin_amdgcn_permlane16(u, u, s0, s1, false, false));
}
__device__ __forceinline__ float redmax16(float v) {
    v = fmaxf(v, permx16(v, 0x67452301u, 0xEFCDAB89u));   // ^1
    v = fmaxf(v, permx16(v, 0x54761032u, 0xDCFE98BAu));   // ^2
    v = fmaxf(v, permx16(v, 0x32107654u, 0xBA98FEDCu));   // ^4
    v = fmaxf(v, permx16(v, 0xFEDCBA98u, 0x76543210u));   // ^8
    return v;
}
__device__ __forceinline__ float redsum16(float v) {
    v += permx16(v, 0x67452301u, 0xEFCDAB89u);
    v += permx16(v, 0x54761032u, 0xDCFE98BAu);
    v += permx16(v, 0x32107654u, 0xBA98FEDCu);
    v += permx16(v, 0xFEDCBA98u, 0x76543210u);
    return v;
}

__device__ __forceinline__ float gelu_tanh(float x) {
    float x3 = x * x * x;
    return 0.5f * x * (1.f + tanhf(0.7978845608f * (x + 0.044715f * x3)));
}

// ---------------------------------------------------------------------------
// fp32 -> bf16 cast (one-time weight conversion)
// ---------------------------------------------------------------------------
__global__ void cast_f32_bf16(const float* __restrict__ s, bf16_t* __restrict__ d, int n) {
    int i = blockIdx.x * 256 + threadIdx.x;
    if (i < n) d[i] = (bf16_t)s[i];
}

// ---------------------------------------------------------------------------
// Fused LayerNorm -> bf16 row (one block per token row, C = 1024)
// ---------------------------------------------------------------------------
__global__ __launch_bounds__(256)
void ln_bf16(const float* __restrict__ x, const float* __restrict__ g,
             const float* __restrict__ bb, bf16_t* __restrict__ out) {
    const int row = blockIdx.x;
    const int tid = threadIdx.x;
    const float4 v = ((const float4*)(x + (size_t)row * CMODEL))[tid];
    __shared__ float rs[256], rq[256];
    rs[tid] = v.x + v.y + v.z + v.w;
    rq[tid] = v.x * v.x + v.y * v.y + v.z * v.z + v.w * v.w;
    __syncthreads();
    for (int off = 128; off > 0; off >>= 1) {
        if (tid < off) { rs[tid] += rs[tid + off]; rq[tid] += rq[tid + off]; }
        __syncthreads();
    }
    const float mu  = rs[0] * (1.f / CMODEL);
    const float var = rq[0] * (1.f / CMODEL) - mu * mu;
    const float inv = rsqrtf(var + 1e-5f);
    const int c0 = tid * 4;
    float vv[4] = {v.x, v.y, v.z, v.w};
#pragma unroll
    for (int i = 0; i < 4; ++i)
        out[(size_t)row * CMODEL + c0 + i] =
            (bf16_t)((vv[i] - mu) * inv * g[c0 + i] + bb[c0 + i]);
}

// ---------------------------------------------------------------------------
// Tiled WMMA GEMM:  out[M,N] = A_bf16[M,K] * W_bf16[N,K]^T (+ bias, epilogue)
// Block: 256 threads (8 waves, 2x4 grid). Tile 128x128, K-step 32.
// Tiles staged by the Tensor Data Mover, double-buffered: wave0 issues the A
// descriptor, wave1 the B descriptor for tile i+1 while all waves run WMMA on
// tile i; s_wait_tensorcnt + barrier closes the pipeline stage.
// EPI 0: fp32 + bias | EPI 1: fp32 + bias + residual | EPI 2: bf16 + bias + GELU
// ---------------------------------------------------------------------------
template <int EPI>
__global__ __launch_bounds__(256)
void gemm_bf16_wmma(const bf16_t* __restrict__ A, const bf16_t* __restrict__ W,
                    const float* __restrict__ bias, const float* __restrict__ resid,
                    float* __restrict__ outF, bf16_t* __restrict__ outB,
                    int M, int N, int K) {
    __shared__ __align__(16) bf16_t As[2][128 * 32];
    __shared__ __align__(16) bf16_t Bs[2][128 * 32];
    const int tid  = threadIdx.x;
    const int lane = tid & 31, wid = tid >> 5;
    const int wm = wid >> 2, wn = wid & 3;       // 2 x 4 wave grid
    const int lo = (lane < 16) ? 0 : 8;
    const int rsel = lane & 15;

    v8f acc[4][2];
#pragma unroll
    for (int mi = 0; mi < 4; ++mi)
#pragma unroll
        for (int ni = 0; ni < 2; ++ni)
#pragma unroll
            for (int e = 0; e < 8; ++e) acc[mi][ni][e] = 0.f;

    const int mBlk = blockIdx.y * 128, nBlk = blockIdx.x * 128;
    const bf16_t* Abase = A + (size_t)mBlk * K;
    const bf16_t* Bbase = W + (size_t)nBlk * K;
    const unsigned ldsA[2] = { lds_addr_of(&As[0][0]), lds_addr_of(&As[1][0]) };
    const unsigned ldsB[2] = { lds_addr_of(&Bs[0][0]), lds_addr_of(&Bs[1][0]) };

    // prologue: TDM-stage tile 0 into buffer 0
    if (wid == 0) tdm_load_2d(Abase, ldsA[0], 32, 128, K);
    if (wid == 1) tdm_load_2d(Bbase, ldsB[0], 32, 128, K);
    if (wid < 2) __builtin_amdgcn_s_wait_tensorcnt(0);
    __syncthreads();

    int p = 0;
    for (int k0 = 0; k0 < K; k0 += 32) {
        const int kn = k0 + 32;
        if (kn < K) {                       // prefetch next tile into other buffer
            if (wid == 0) tdm_load_2d(Abase + kn, ldsA[p ^ 1], 32, 128, K);
            if (wid == 1) tdm_load_2d(Bbase + kn, ldsB[p ^ 1], 32, 128, K);
        }
        v16bf bfrag[2];
#pragma unroll
        for (int ni = 0; ni < 2; ++ni)
            bfrag[ni] = load_frag16(&Bs[p][(wn * 32 + ni * 16 + rsel) * 32], lo);
#pragma unroll
        for (int mi = 0; mi < 4; ++mi) {
            v16bf afrag = load_frag16(&As[p][(wm * 64 + mi * 16 + rsel) * 32], lo);
#pragma unroll
            for (int ni = 0; ni < 2; ++ni)
                acc[mi][ni] = __builtin_amdgcn_wmma_f32_16x16x32_bf16(
                    false, afrag, false, bfrag[ni], (short)0, acc[mi][ni], false, false);
        }
        if (wid < 2) __builtin_amdgcn_s_wait_tensorcnt(0);
        __syncthreads();
        p ^= 1;
    }

    const int hhalf = lane >> 4;
#pragma unroll
    for (int mi = 0; mi < 4; ++mi)
#pragma unroll
        for (int ni = 0; ni < 2; ++ni)
#pragma unroll
            for (int r = 0; r < 8; ++r) {
                const int gm = mBlk + wm * 64 + mi * 16 + r + 8 * hhalf;
                const int gn = nBlk + wn * 32 + ni * 16 + rsel;
                const size_t idx = (size_t)gm * N + gn;
                float v = acc[mi][ni][r] + bias[gn];
                if (EPI == 0)       outF[idx] = v;
                else if (EPI == 1)  outF[idx] = v + resid[idx];
                else                outB[idx] = (bf16_t)gelu_tanh(v);
            }
}

// ---------------------------------------------------------------------------
// RoPE + head split:  qkv fp32 (row = t*B+b, 3C) -> bf16 Q/K/V in (B*H, T, D)
// Q is pre-scaled by D^-0.5 = 1/8.
// ---------------------------------------------------------------------------
__global__ __launch_bounds__(256)
void rope_split(const float* __restrict__ qkv, bf16_t* __restrict__ qo,
                bf16_t* __restrict__ ko, bf16_t* __restrict__ vo) {
    const int row = blockIdx.x;            // t*B + b
    const int t = row >> 1, b = row & 1;
    const float* base = qkv + (size_t)row * 3 * CMODEL;
    for (int idx = threadIdx.x; idx < CMODEL; idx += 256) {
        const int h = idx >> 6, d = idx & 63;
        const int dd = d & 31;
        const float theta = __powf(10000.f, -(float)(2 * dd) * (1.f / 64.f));
        const float mt = (float)t * theta;
        const float cs = __cosf(mt), sn = __sinf(mt);
        const float qv = base[idx];
        const float kv = base[CMODEL + idx];
        const float vv = base[2 * CMODEL + idx];
        const int pidx = (h << 6) | (d ^ 32);
        const float q2 = base[pidx];
        const float k2 = base[CMODEL + pidx];
        const float qr = (d < 32) ? (cs * qv - sn * q2) : (cs * qv + sn * q2);
        const float kr = (d < 32) ? (cs * kv - sn * k2) : (cs * kv + sn * k2);
        const size_t o = ((size_t)(b * NHEAD + h) * T_SEQ + t) * DHEAD + d;
        qo[o] = (bf16_t)(qr * 0.125f);
        ko[o] = (bf16_t)kr;
        vo[o] = (bf16_t)vv;
    }
}

// ---------------------------------------------------------------------------
// Fused causal flash attention (bf16 WMMA, fp32 softmax state).
// Grid: (T/128, B*H). Block: 256 threads = 8 waves; wave owns 16 Q rows.
// K chunk staged by TDM; V chunk transposed manually into LDS.
// ---------------------------------------------------------------------------
__global__ __launch_bounds__(256)
void attn_fwd(const bf16_t* __restrict__ Qp, const bf16_t* __restrict__ Kp,
              const bf16_t* __restrict__ Vp, bf16_t* __restrict__ Op) {
    __shared__ __align__(16) bf16_t Ks[32 * 64];       // [s][d]
    __shared__ __align__(16) bf16_t Vt[64 * 32];       // [d][s]  (transposed)
    __shared__ __align__(16) bf16_t Ps[8 * 16 * 32];   // per-wave P tile
    const int bh = blockIdx.y;
    const int mbase = blockIdx.x * 128;
    const int tid = threadIdx.x, lane = tid & 31, wid = tid >> 5;
    const int lo = (lane < 16) ? 0 : 8;
    const int rsel = lane & 15;
    const int hhalf = lane >> 4;
    const int wrow = mbase + wid * 16;                 // this wave's Q rows
    const unsigned ldsK = lds_addr_of(&Ks[0]);

    // Q fragments stay in registers for the whole pass
    v16bf qf[2];
    const bf16_t* Qb = Qp + ((size_t)bh * T_SEQ + wrow) * DHEAD;
#pragma unroll
    for (int kk = 0; kk < 2; ++kk)
        qf[kk] = load_frag16(&Qb[rsel * DHEAD + kk * 32], lo);

    float mrow[8], lrow[8];
    v8f oacc[4];
#pragma unroll
    for (int r = 0; r < 8; ++r) { mrow[r] = -1e9f; lrow[r] = 0.f; }
#pragma unroll
    for (int ni = 0; ni < 4; ++ni)
#pragma unroll
        for (int e = 0; e < 8; ++e) oacc[ni][e] = 0.f;

    const int ldlin = tid * 8;
    const int lsr = ldlin >> 6;        // s row in chunk 0..31
    const int ldc = ldlin & 63;        // d column

    for (int s0 = 0; s0 < mbase + 128; s0 += 32) {
        __syncthreads();
        // K chunk: 32 contiguous rows of 64 -> one TDM descriptor from wave 0
        if (wid == 0)
            tdm_load_2d(Kp + ((size_t)bh * T_SEQ + s0) * DHEAD, ldsK, 64, 32, 64);
        // V chunk transposed into Vt[d][s] by all threads
        bf16x8 vv = *(const bf16x8*)&Vp[((size_t)bh * T_SEQ + s0 + lsr) * DHEAD + ldc];
#pragma unroll
        for (int j = 0; j < 8; ++j) Vt[(ldc + j) * 32 + lsr] = vv[j];
        if (wid == 0) __builtin_amdgcn_s_wait_tensorcnt(0);
        __syncthreads();

        // scores: Q(16x64) * K^T(64x32)  -> two 16x16 tiles
        v8f sacc[2];
#pragma unroll
        for (int j = 0; j < 2; ++j)
#pragma unroll
            for (int e = 0; e < 8; ++e) sacc[j][e] = 0.f;
#pragma unroll
        for (int kk = 0; kk < 2; ++kk)
#pragma unroll
            for (int j = 0; j < 2; ++j) {
                v16bf bf = load_frag16(&Ks[(j * 16 + rsel) * 64 + kk * 32], lo);
                sacc[j] = __builtin_amdgcn_wmma_f32_16x16x32_bf16(
                    false, qf[kk], false, bf, (short)0, sacc[j], false, false);
            }

        // causal mask + chunk max (select, not branch: EXEC stays full)
        float cmax[8], csum[8];
#pragma unroll
        for (int r = 0; r < 8; ++r) { cmax[r] = -1e9f; csum[r] = 0.f; }
#pragma unroll
        for (int j = 0; j < 2; ++j) {
            const int col = s0 + j * 16 + rsel;
#pragma unroll
            for (int r = 0; r < 8; ++r) {
                const int grow = wrow + r + 8 * hhalf;
                float v = sacc[j][r];
                v = (col > grow) ? -1e9f : v;
                sacc[j][r] = v;
                cmax[r] = fmaxf(cmax[r], v);
            }
        }
        // online softmax update; rows live across 16 lanes of each half-wave
#pragma unroll
        for (int r = 0; r < 8; ++r) {
            cmax[r] = redmax16(cmax[r]);
            const float mn  = fmaxf(mrow[r], cmax[r]);
            const float fac = __expf(mrow[r] - mn);
            mrow[r] = mn;
            lrow[r] *= fac;
#pragma unroll
            for (int ni = 0; ni < 4; ++ni) oacc[ni][r] *= fac;
        }
#pragma unroll
        for (int j = 0; j < 2; ++j)
#pragma unroll
            for (int r = 0; r < 8; ++r) {
                const float pp = __expf(sacc[j][r] - mrow[r]);
                sacc[j][r] = pp;
                csum[r] += pp;
            }
#pragma unroll
        for (int r = 0; r < 8; ++r)
            lrow[r] += redsum16(csum[r]);

        // P: C-layout -> A-layout via per-wave LDS round trip
        bf16_t* Pw = &Ps[wid * 512];
#pragma unroll
        for (int j = 0; j < 2; ++j)
#pragma unroll
            for (int r = 0; r < 8; ++r)
                Pw[(r + 8 * hhalf) * 32 + j * 16 + rsel] = (bf16_t)sacc[j][r];
        v16bf pf = load_frag16(&Pw[rsel * 32], lo);

        // O += P(16x32) * V(32x64): 4 N-tiles of 16 (Vt rows contiguous in K)
#pragma unroll
        for (int ni = 0; ni < 4; ++ni) {
            v16bf vf = load_frag16(&Vt[(ni * 16 + rsel) * 32], lo);
            oacc[ni] = __builtin_amdgcn_wmma_f32_16x16x32_bf16(
                false, pf, false, vf, (short)0, oacc[ni], false, false);
        }
    }

    // finalize: divide by row sum, scatter to (T, B, C) bf16
    const int b = bh >> 4, h = bh & 15;
#pragma unroll
    for (int ni = 0; ni < 4; ++ni)
#pragma unroll
        for (int r = 0; r < 8; ++r) {
            const int t = wrow + r + 8 * hhalf;
            const int c = h * DHEAD + ni * 16 + rsel;
            Op[((size_t)t * BATCH + b) * CMODEL + c] = (bf16_t)(oacc[ni][r] / lrow[r]);
        }
}

// ---------------------------------------------------------------------------
// Launcher
// ---------------------------------------------------------------------------
extern "C" void kernel_launch(void* const* d_in, const int* in_sizes, int n_in,
                              void* d_out, int out_size, void* d_ws, size_t ws_size,
                              hipStream_t stream) {
    (void)in_sizes; (void)n_in; (void)out_size; (void)ws_size;
    const float* x     = (const float*)d_in[0];
    const float* in_w  = (const float*)d_in[1];
    const float* in_b  = (const float*)d_in[2];
    const float* out_w = (const float*)d_in[3];
    const float* out_b = (const float*)d_in[4];
    const float* ln1_g = (const float*)d_in[5];
    const float* ln1_b = (const float*)d_in[6];
    const float* ln2_g = (const float*)d_in[7];
    const float* ln2_b = (const float*)d_in[8];
    const float* w1    = (const float*)d_in[9];
    const float* b1    = (const float*)d_in[10];
    const float* w2    = (const float*)d_in[11];
    const float* b2    = (const float*)d_in[12];

    char* ws = (char*)d_ws;
    bf16_t* wbf_in  = (bf16_t*)(ws + 0);           // 3072x1024 bf16
    bf16_t* wbf_out = (bf16_t*)(ws + 6291456);     // 1024x1024 bf16
    bf16_t* wbf_w1  = (bf16_t*)(ws + 8388608);     // 4096x1024 bf16
    bf16_t* wbf_w2  = (bf16_t*)(ws + 16777216);    // 1024x4096 bf16
    bf16_t* h1_bf   = (bf16_t*)(ws + 25165824);    // 4096x1024 bf16
    float*  qkv     = (float*) (ws + 33554432);    // 4096x3072 f32
    bf16_t* q_bf    = (bf16_t*)(ws + 83886080);    // (32,2048,64) bf16
    bf16_t* k_bf    = (bf16_t*)(ws + 92274688);
    bf16_t* v_bf    = (bf16_t*)(ws + 100663296);
    bf16_t* o_bf    = (bf16_t*)(ws + 109051904);   // 4096x1024 bf16
    float*  xmid    = (float*) (ws + 117440512);   // 4096x1024 f32
    bf16_t* h2_bf   = (bf16_t*)(ws + 134217728);   // 4096x1024 bf16
    bf16_t* ffn1_bf = (bf16_t*)(ws + 142606336);   // 4096x4096 bf16
    float*  out     = (float*)d_out;               // 4096x1024 f32

    // 1. one-time weight casts
    cast_f32_bf16<<<(3 * CMODEL * CMODEL) / 256, 256, 0, stream>>>(in_w, wbf_in, 3 * CMODEL * CMODEL);
    cast_f32_bf16<<<(CMODEL * CMODEL) / 256, 256, 0, stream>>>(out_w, wbf_out, CMODEL * CMODEL);
    cast_f32_bf16<<<(FDIM * CMODEL) / 256, 256, 0, stream>>>(w1, wbf_w1, FDIM * CMODEL);
    cast_f32_bf16<<<(CMODEL * FDIM) / 256, 256, 0, stream>>>(w2, wbf_w2, CMODEL * FDIM);

    // 2. LN1 -> bf16
    ln_bf16<<<MROWS, 256, 0, stream>>>(x, ln1_g, ln1_b, h1_bf);

    // 3. QKV projection (4096 x 3072 x 1024)
    gemm_bf16_wmma<0><<<dim3(3 * CMODEL / 128, MROWS / 128), 256, 0, stream>>>(
        h1_bf, wbf_in, in_b, nullptr, qkv, nullptr, MROWS, 3 * CMODEL, CMODEL);

    // 4. RoPE + head split
    rope_split<<<MROWS, 256, 0, stream>>>(qkv, q_bf, k_bf, v_bf);

    // 5. fused causal flash attention
    attn_fwd<<<dim3(T_SEQ / 128, BATCH * NHEAD), 256, 0, stream>>>(q_bf, k_bf, v_bf, o_bf);

    // 6. out projection + residual (4096 x 1024 x 1024)
    gemm_bf16_wmma<1><<<dim3(CMODEL / 128, MROWS / 128), 256, 0, stream>>>(
        o_bf, wbf_out, out_b, x, xmid, nullptr, MROWS, CMODEL, CMODEL);

    // 7. LN2 -> bf16
    ln_bf16<<<MROWS, 256, 0, stream>>>(xmid, ln2_g, ln2_b, h2_bf);

    // 8. FFN1 + GELU (4096 x 4096 x 1024) -> bf16
    gemm_bf16_wmma<2><<<dim3(FDIM / 128, MROWS / 128), 256, 0, stream>>>(
        h2_bf, wbf_w1, b1, nullptr, nullptr, ffn1_bf, MROWS, FDIM, CMODEL);

    // 9. FFN2 + residual (4096 x 1024 x 4096) -> fp32 final output
    gemm_bf16_wmma<1><<<dim3(CMODEL / 128, MROWS / 128), 256, 0, stream>>>(
        ffn1_bf, wbf_w2, b2, xmid, out, nullptr, MROWS, CMODEL, FDIM);
}
